// Model_77807627534751
// MI455X (gfx1250) — matrix operand order repaired and checked by reading, not compiled, not run on hardware
//
#include <hip/hip_runtime.h>
#include <hip/hip_bf16.h>

// ---------------- problem constants (from reference) ----------------
#define NATOMS   4096
#define NTYPES   2
#define DDESC    352          // 2*128 radial + 3*2*16 angular
#define SELJ     128
#define ASEL     16
#define HDIM     256
#define EPSF     1e-16f

typedef __attribute__((ext_vector_type(16))) _Float16 v16h;
typedef __attribute__((ext_vector_type(8)))  _Float16 v8h;
typedef __attribute__((ext_vector_type(8)))  float    v8f;

// =====================================================================
// Kernel 1: descriptors.  One 128-thread block per atom i.
// LDS bitonic sort of 4096 (dist,index) keys per type -> sorted top-k.
// =====================================================================
__global__ __launch_bounds__(128) void desc_kernel(
    const float* __restrict__ xyz,     // [NATOMS][3]
    const float* __restrict__ box,     // [3]
    const int*   __restrict__ types,   // [NATOMS]
    _Float16*    __restrict__ desc)    // [NATOMS][DDESC] f16
{
    __shared__ unsigned long long keys[NATOMS];   // 32 KB
    __shared__ float nbrD[2 * ASEL];
    __shared__ int   nbrI[2 * ASEL];
    __shared__ float lfD[4];
    __shared__ int   lfI[4];
    __shared__ float Af[9];                       // frame rows r0, v2, v3

    const int i   = blockIdx.x;
    const int tid = threadIdx.x;

    const float bx = box[0], by = box[1], bz = box[2];
    const float xi = xyz[3 * i + 0];
    const float yi = xyz[3 * i + 1];
    const float zi = xyz[3 * i + 2];

    for (int t = 0; t < NTYPES; ++t) {
        // ---- fill keys: masked distances ----
        for (int j = tid; j < NATOMS; j += 128) {
            float dx = xi - xyz[3 * j + 0] + EPSF; dx -= bx * rintf(dx / bx);
            float dy = yi - xyz[3 * j + 1] + EPSF; dy -= by * rintf(dy / by);
            float dz = zi - xyz[3 * j + 2] + EPSF; dz -= bz * rintf(dz / bz);
            float d  = sqrtf(dx * dx + dy * dy + dz * dz);
            bool ok  = (types[j] == t) && (j != i);
            unsigned int bits = ok ? __float_as_uint(d) : 0x7F800000u; // +inf
            keys[j] = ((unsigned long long)bits << 32) | (unsigned int)j;
        }
        __syncthreads();

        // ---- bitonic sort ascending (positive float bits are monotone) ----
        for (int size = 2; size <= NATOMS; size <<= 1) {
            for (int stride = size >> 1; stride > 0; stride >>= 1) {
                for (int idx = tid; idx < NATOMS / 2; idx += 128) {
                    int lo = ((idx / stride) * (stride << 1)) + (idx % stride);
                    int hi = lo + stride;
                    bool asc = ((lo & size) == 0);
                    unsigned long long a = keys[lo];
                    unsigned long long b = keys[hi];
                    if ((a > b) == asc) { keys[lo] = b; keys[hi] = a; }
                }
                __syncthreads();
            }
        }

        // ---- radial block: 1/(d+eps) for the 128 nearest of this type ----
        {
            unsigned long long k = keys[tid];
            float d = __uint_as_float((unsigned int)(k >> 32));
            desc[(size_t)i * DDESC + t * SELJ + tid] = (_Float16)(1.0f / (d + EPSF));
        }
        if (tid < ASEL) {
            unsigned long long k = keys[tid];
            nbrD[t * ASEL + tid] = __uint_as_float((unsigned int)(k >> 32));
            nbrI[t * ASEL + tid] = (int)(unsigned int)(k & 0xFFFFFFFFull);
        }
        if (tid < 2) {
            unsigned long long k = keys[tid];
            lfD[t * 2 + tid] = __uint_as_float((unsigned int)(k >> 32));
            lfI[t * 2 + tid] = (int)(unsigned int)(k & 0xFFFFFFFFull);
        }
        __syncthreads();
    }

    // ---- local frame (two globally nearest of the 4 candidates, stable) ----
    if (tid == 0) {
        int b0 = 0;
        for (int c = 1; c < 4; ++c) if (lfD[c] < lfD[b0]) b0 = c;
        int b1 = -1;
        for (int c = 0; c < 4; ++c) {
            if (c == b0) continue;
            if (b1 < 0 || lfD[c] < lfD[b1]) b1 = c;
        }
        int j0 = lfI[b0], j1 = lfI[b1];
        float d0 = lfD[b0], d1 = lfD[b1];

        float r0x = xi - xyz[3 * j0 + 0] + EPSF; r0x -= bx * rintf(r0x / bx);
        float r0y = yi - xyz[3 * j0 + 1] + EPSF; r0y -= by * rintf(r0y / by);
        float r0z = zi - xyz[3 * j0 + 2] + EPSF; r0z -= bz * rintf(r0z / bz);
        float inv0 = 1.0f / (d0 + EPSF);
        r0x *= inv0; r0y *= inv0; r0z *= inv0;

        float r1x = xi - xyz[3 * j1 + 0] + EPSF; r1x -= bx * rintf(r1x / bx);
        float r1y = yi - xyz[3 * j1 + 1] + EPSF; r1y -= by * rintf(r1y / by);
        float r1z = zi - xyz[3 * j1 + 2] + EPSF; r1z -= bz * rintf(r1z / bz);
        float inv1 = 1.0f / (d1 + EPSF);
        r1x *= inv1; r1y *= inv1; r1z *= inv1;

        float dot = r0x * r1x + r0y * r1y + r0z * r1z;
        float v2x = r1x - dot * r0x;
        float v2y = r1y - dot * r0y;
        float v2z = r1z - dot * r0z;
        float n2 = rsqrtf(v2x * v2x + v2y * v2y + v2z * v2z);
        v2x *= n2; v2y *= n2; v2z *= n2;

        float v3x = r0y * r1z - r0z * r1y;
        float v3y = r0z * r1x - r0x * r1z;
        float v3z = r0x * r1y - r0y * r1x;
        float n3 = rsqrtf(v3x * v3x + v3y * v3y + v3z * v3z);
        v3x *= n3; v3y *= n3; v3z *= n3;

        Af[0] = r0x; Af[1] = r0y; Af[2] = r0z;
        Af[3] = v2x; Af[4] = v2y; Af[5] = v2z;
        Af[6] = v3x; Af[7] = v3y; Af[8] = v3z;
    }
    __syncthreads();

    // ---- angular block: rot[k][c] = (A[c] . dm_k) / (d_k+eps)^2 ----
    if (tid < 2 * ASEL) {
        int k = tid;
        int j = nbrI[k];
        float d = nbrD[k];
        float dx = xi - xyz[3 * j + 0] + EPSF; dx -= bx * rintf(dx / bx);
        float dy = yi - xyz[3 * j + 1] + EPSF; dy -= by * rintf(dy / by);
        float dz = zi - xyz[3 * j + 2] + EPSF; dz -= bz * rintf(dz / bz);
        float invd2 = 1.0f / ((d + EPSF) * (d + EPSF));
#pragma unroll
        for (int c = 0; c < 3; ++c) {
            float val = (Af[3 * c + 0] * dx + Af[3 * c + 1] * dy + Af[3 * c + 2] * dz) * invd2;
            desc[(size_t)i * DDESC + 2 * SELJ + k * 3 + c] = (_Float16)val;
        }
    }
}

// =====================================================================
// Kernel 2: weight convert + transpose: f32 [T][K][N] -> f16 [T][N][K]
// (makes the WMMA B-fragment a contiguous 32B per-lane load)
// =====================================================================
__global__ void wt_transpose_kernel(const float* __restrict__ src,
                                    _Float16* __restrict__ dst,
                                    int K, int Nout)
{
    int idx = blockIdx.x * blockDim.x + threadIdx.x;
    int total = NTYPES * K * Nout;
    if (idx >= total) return;
    int t = idx / (K * Nout);
    int r = idx - t * (K * Nout);
    int k = r / Nout;
    int n = r - k * Nout;
    dst[((size_t)t * Nout + n) * K + k] = (_Float16)src[idx];
}

// =====================================================================
// Kernel 3: WMMA MLP layer.  One wave computes a 16x64 output tile
// (4 N-tiles sharing one A fragment).  K is compile-time -> full unroll.
//   Y[t][m][n] = act( sum_k X[(t)][m][k] * W[t][k][n] + B[t][n] )
// grid = (NATOMS/16, Nout/64, NTYPES), block = 32 (EXEC all ones).
// =====================================================================
template <int K, int XPT, int TANH>
__global__ __launch_bounds__(32) void mlp_layer_kernel(
    const _Float16* __restrict__ X,   // [Nat][K] or [T][Nat][K]
    const _Float16* __restrict__ WT,  // [T][Nout][K]  (transposed f16)
    const float*    __restrict__ B,   // [T][Nout]
    _Float16*       __restrict__ Y,   // [T][Nat][Nout]
    int Nout)
{
    const int mt = blockIdx.x;        // M tile (16 atoms)
    const int ng = blockIdx.y;        // group of 4 N tiles (64 outputs)
    const int t  = blockIdx.z;
    const int lane = threadIdx.x;
    const int l  = lane & 15;
    const int hh = lane >> 4;

    const _Float16* Xt   = X + (XPT ? (size_t)t * NATOMS * K : (size_t)0);
    const _Float16* arow = Xt + (size_t)(mt * 16 + l) * K;

    // per-lane B rows (transposed weights): lane l owns column n of each tile
    const _Float16* brow0 = WT + ((size_t)t * Nout + (ng * 64 +  0 + l)) * K;
    const _Float16* brow1 = WT + ((size_t)t * Nout + (ng * 64 + 16 + l)) * K;
    const _Float16* brow2 = WT + ((size_t)t * Nout + (ng * 64 + 32 + l)) * K;
    const _Float16* brow3 = WT + ((size_t)t * Nout + (ng * 64 + 48 + l)) * K;

    v8f acc0 = {}, acc1 = {}, acc2 = {}, acc3 = {};

#pragma unroll
    for (int k0 = 0; k0 < K; k0 += 32) {
        // A fragment: lane group hh holds K {16hh..16hh+15}
        // -> two contiguous 16B loads
        v8h a0 = *(const v8h*)(arow + k0 + 16 * hh);
        v8h a1 = *(const v8h*)(arow + k0 + 16 * hh + 8);
        v16h a = __builtin_shufflevector(a0, a1,
                 0, 1, 2, 3, 4, 5, 6, 7, 8, 9, 10, 11, 12, 13, 14, 15);

        // B fragment: lane group hh holds K {16hh..16hh+15} of column n
        // -> one contiguous 32B load per tile
        const int kb = k0 + 16 * hh;
        v16h b0 = *(const v16h*)(brow0 + kb);
        v16h b1 = *(const v16h*)(brow1 + kb);
        v16h b2 = *(const v16h*)(brow2 + kb);
        v16h b3 = *(const v16h*)(brow3 + kb);

        acc0 = __builtin_amdgcn_wmma_f32_16x16x32_f16(false, a, false, b0,
                                                      (short)0, acc0, false, false);
        acc1 = __builtin_amdgcn_wmma_f32_16x16x32_f16(false, a, false, b1,
                                                      (short)0, acc1, false, false);
        acc2 = __builtin_amdgcn_wmma_f32_16x16x32_f16(false, a, false, b2,
                                                      (short)0, acc2, false, false);
        acc3 = __builtin_amdgcn_wmma_f32_16x16x32_f16(false, a, false, b3,
                                                      (short)0, acc3, false, false);
    }

    // D layout: lane l -> column n; VGPR r -> row m = r + 8*hh
    v8f* accs[4] = { &acc0, &acc1, &acc2, &acc3 };
#pragma unroll
    for (int q = 0; q < 4; ++q) {
        const int n = ng * 64 + q * 16 + l;
        const float bias = B[t * Nout + n];
        v8f& acc = *accs[q];
#pragma unroll
        for (int r = 0; r < 8; ++r) {
            const int mm = mt * 16 + r + 8 * hh;
            float val = acc[r] + bias;
            if (TANH) val = tanhf(val);
            Y[((size_t)t * NATOMS + mm) * Nout + n] = (_Float16)val;
        }
    }
}

// =====================================================================
// Kernel 4: final layer (256 -> 1) + per-atom type gather, f32 precise
// =====================================================================
__global__ __launch_bounds__(256) void energy_kernel(
    const _Float16* __restrict__ H,   // [T][NATOMS][HDIM]
    const float*    __restrict__ w4,  // [T][HDIM][1]
    const float*    __restrict__ b4,  // [T][1]
    const int*      __restrict__ types,
    float*          __restrict__ partial) // [NATOMS]
{
    int n = blockIdx.x * blockDim.x + threadIdx.x;
    if (n >= NATOMS) return;
    int t = types[n];
    const _Float16* h = H + ((size_t)t * NATOMS + n) * HDIM;
    const float* w = w4 + (size_t)t * HDIM;
    float s = 0.0f;
#pragma unroll
    for (int f = 0; f < HDIM; f += 8) {
        v8h hv = *(const v8h*)(h + f);
#pragma unroll
        for (int j = 0; j < 8; ++j) s += (float)hv[j] * w[f + j];
    }
    partial[n] = s + b4[t];
}

__global__ __launch_bounds__(256) void reduce_kernel(
    const float* __restrict__ partial, float* __restrict__ out)
{
    __shared__ float sm[256];
    float s = 0.0f;
    for (int idx = threadIdx.x; idx < NATOMS; idx += 256) s += partial[idx];
    sm[threadIdx.x] = s;
    __syncthreads();
    for (int off = 128; off > 0; off >>= 1) {
        if ((int)threadIdx.x < off) sm[threadIdx.x] += sm[threadIdx.x + off];
        __syncthreads();
    }
    if (threadIdx.x == 0) out[0] = sm[0];
}

// =====================================================================
// Host side
// =====================================================================
extern "C" void kernel_launch(void* const* d_in, const int* in_sizes, int n_in,
                              void* d_out, int out_size, void* d_ws, size_t ws_size,
                              hipStream_t stream)
{
    const float* xyz   = (const float*)d_in[0];
    const float* box   = (const float*)d_in[1];
    const int*   types = (const int*)  d_in[2];
    const float* w1 = (const float*)d_in[3];
    const float* b1 = (const float*)d_in[4];
    const float* w2 = (const float*)d_in[5];
    const float* b2 = (const float*)d_in[6];
    const float* w3 = (const float*)d_in[7];
    const float* b3 = (const float*)d_in[8];
    const float* w4 = (const float*)d_in[9];
    const float* b4 = (const float*)d_in[10];

    // ---- workspace carve-up (256B aligned) ----
    char* ws = (char*)d_ws;
    size_t off = 0;
    auto carve = [&](size_t bytes) -> void* {
        void* p = ws + off;
        off += (bytes + 255) & ~(size_t)255;
        return p;
    };
    const int nW1 = NTYPES * DDESC * HDIM;  // 180224
    const int nW2 = NTYPES * HDIM * HDIM;   // 131072
    const int nW3 = NTYPES * HDIM * HDIM;   // 131072

    _Float16* descF = (_Float16*)carve((size_t)NATOMS * DDESC * sizeof(_Float16));
    _Float16* hA    = (_Float16*)carve((size_t)NTYPES * NATOMS * HDIM * sizeof(_Float16));
    _Float16* hB    = (_Float16*)carve((size_t)NTYPES * NATOMS * HDIM * sizeof(_Float16));
    _Float16* w1t   = (_Float16*)carve((size_t)nW1 * sizeof(_Float16));
    _Float16* w2t   = (_Float16*)carve((size_t)nW2 * sizeof(_Float16));
    _Float16* w3t   = (_Float16*)carve((size_t)nW3 * sizeof(_Float16));
    float*    part  = (float*)   carve((size_t)NATOMS * sizeof(float));

    // ---- weight conversion + transpose ([T][K][N] -> [T][N][K] f16) ----
    wt_transpose_kernel<<<(nW1 + 255) / 256, 256, 0, stream>>>(w1, w1t, DDESC, HDIM);
    wt_transpose_kernel<<<(nW2 + 255) / 256, 256, 0, stream>>>(w2, w2t, HDIM, HDIM);
    wt_transpose_kernel<<<(nW3 + 255) / 256, 256, 0, stream>>>(w3, w3t, HDIM, HDIM);

    // ---- descriptors ----
    desc_kernel<<<NATOMS, 128, 0, stream>>>(xyz, box, types, descF);

    // ---- WMMA MLP: 352->256->256->256 with tanh ----
    dim3 g(NATOMS / 16, HDIM / 64, NTYPES);
    mlp_layer_kernel<DDESC, 0, 1><<<g, 32, 0, stream>>>(descF, w1t, b1, hA, HDIM);
    mlp_layer_kernel<HDIM,  1, 1><<<g, 32, 0, stream>>>(hA,    w2t, b2, hB, HDIM);
    mlp_layer_kernel<HDIM,  1, 1><<<g, 32, 0, stream>>>(hB,    w3t, b3, hA, HDIM);

    // ---- energies + deterministic reduction ----
    energy_kernel<<<NATOMS / 256, 256, 0, stream>>>(hA, w4, b4, types, part);
    reduce_kernel<<<1, 256, 0, stream>>>(part, (float*)d_out);
}
